// MultiHeadAttention_38173669326938
// MI455X (gfx1250) — matrix operand order repaired and checked by reading, not compiled, not run on hardware
//
#include <hip/hip_runtime.h>
#include <hip/hip_bf16.h>

// ---------------------------------------------------------------------------
// MI455X (gfx1250, wave32) multi-head attention, all GEMMs via
// v_wmma_f32_16x16x32_bf16.  B=2, S=2048, D=1024, H=16, DH=64.
//
// Pipeline:
//   0) k_transpose_cvt : fp32 weights -> bf16, transposed (B-operand layout)
//   1) k_qkv_proj      : per-head QKV projection (V stored transposed)
//   2) k_attn          : flash-style causal attention (online softmax)
//   3) k_oproj         : output projection + bias (fp32 out)
// All B-fragments are two 16B loads per lane (global bf16, L2-resident);
// no LDS staging / barriers in the GEMM kernels.
// ---------------------------------------------------------------------------

#define BB 2
#define SS 2048
#define DD 1024
#define HH_HEADS 16
#define DHH 64

typedef __bf16 bf16_t;
typedef __attribute__((ext_vector_type(16))) __bf16 v16bf;
typedef __attribute__((ext_vector_type(8)))  __bf16 v8bf;
typedef __attribute__((ext_vector_type(8)))  float  v8f;

union Frag16 {
    v16bf v;
    v8bf  h[2];
};

static __device__ __forceinline__ v8f wmma_bf16(v16bf a, v16bf b, v8f c) {
    // D = A(16x32 bf16) * B(32x16 bf16) + C(16x16 f32), wave32
    return __builtin_amdgcn_wmma_f32_16x16x32_bf16(
        /*neg_a=*/false, a, /*neg_b=*/false, b,
        /*c_mod=*/(short)0, c, /*reuse_a=*/false, /*reuse_b=*/false);
}

static __device__ __forceinline__ void zero8(v8f& x) {
#pragma unroll
    for (int i = 0; i < 8; ++i) x[i] = 0.0f;
}

// ---------------------------------------------------------------------------
// Kernel 0: tiled transpose + fp32->bf16 convert.
//   in : [M][N] fp32   ->   out : [N][M] bf16        (blockIdx.z: matrix idx)
// 32x32 tiles through LDS, fully coalesced on both sides.
// ---------------------------------------------------------------------------
__global__ void __launch_bounds__(256)
k_transpose_cvt(const float* __restrict__ in, bf16_t* __restrict__ out,
                int M, int N)
{
    const size_t moff = (size_t)blockIdx.z * (size_t)M * (size_t)N;
    in  += moff;
    out += moff;

    __shared__ float tile[32][33];

    const int t  = threadIdx.x;
    const int lr = t >> 5;       // 0..7
    const int lc = t & 31;       // 0..31
    const int c0 = blockIdx.x * 32;
    const int r0 = blockIdx.y * 32;

#pragma unroll
    for (int i = 0; i < 4; ++i)
        tile[lr + i * 8][lc] = in[(size_t)(r0 + lr + i * 8) * N + c0 + lc];
    __syncthreads();
#pragma unroll
    for (int i = 0; i < 4; ++i)
        out[(size_t)(c0 + lr + i * 8) * M + r0 + lc] =
            (bf16_t)tile[lc][lr + i * 8];
}

// ---------------------------------------------------------------------------
// Kernel 1: per-head QKV projection.
//   out[b,h,s,e] = sum_d X[b,s,d] * W[h,d,e] + bias[h,e]
// grid = (rowTiles=32, heads=16, z={q,k,v}), block = 256 (8 waves).
// Each wave computes a 16x64 tile (4 WMMA accumulators); K-loop over D in
// steps of 32.  B-fragments load contiguously from bf16 transposed weights
// WT[h][e][d]; A-fragments load as float4 from global + in-register cvt.
// Q,K stored [B,H,S,DH]; V stored transposed [B,H,DH,S].
// ---------------------------------------------------------------------------
__global__ void __launch_bounds__(256)
k_qkv_proj(const float* __restrict__ xq, const float* __restrict__ xk,
           const float* __restrict__ xv,
           const bf16_t* __restrict__ WqT, const bf16_t* __restrict__ WkT,
           const bf16_t* __restrict__ WvT,
           const float* __restrict__ bq, const float* __restrict__ bk,
           const float* __restrict__ bv,
           bf16_t* __restrict__ Qp, bf16_t* __restrict__ Kp,
           bf16_t* __restrict__ Vt)
{
    const int t    = threadIdx.x;
    const int wave = t >> 5;
    const int lane = t & 31;
    const int nl   = lane & 15;
    const int half = lane >> 4;          // 0: lanes 0-15, 1: lanes 16-31
    const int hh   = half * 8;           // B-fragment K-half offset
    const int h    = blockIdx.y;
    const int z    = blockIdx.z;

    const float*  X    = (z == 0) ? xq : (z == 1) ? xk : xv;
    const bf16_t* WT   = ((z == 0) ? WqT : (z == 1) ? WkT : WvT)
                         + (size_t)h * DHH * DD;               // [e][d]
    const float*  bias = ((z == 0) ? bq : (z == 1) ? bk : bv) + h * DHH;

    const int m0   = blockIdx.x * 128 + wave * 16;   // 16-row strip base
    const int arow = m0 + nl;                        // A-fragment row

    v8f acc[4];
#pragma unroll
    for (int tt = 0; tt < 4; ++tt) zero8(acc[tt]);

#pragma unroll 2
    for (int kc = 0; kc < DD; kc += 32) {
        // A fragment (16x32 bf16): lanes 0-15 K=[0..7,16..23],
        //                          lanes 16-31 K=[8..15,24..31]
        Frag16 a;
        {
            const float* ap = X + (size_t)arow * DD + kc + half * 8;
            const float4 f0 = *(const float4*)(ap);
            const float4 f1 = *(const float4*)(ap + 4);
            const float4 f2 = *(const float4*)(ap + 16);
            const float4 f3 = *(const float4*)(ap + 20);
            a.v[0]  = (bf16_t)f0.x; a.v[1]  = (bf16_t)f0.y;
            a.v[2]  = (bf16_t)f0.z; a.v[3]  = (bf16_t)f0.w;
            a.v[4]  = (bf16_t)f1.x; a.v[5]  = (bf16_t)f1.y;
            a.v[6]  = (bf16_t)f1.z; a.v[7]  = (bf16_t)f1.w;
            a.v[8]  = (bf16_t)f2.x; a.v[9]  = (bf16_t)f2.y;
            a.v[10] = (bf16_t)f2.z; a.v[11] = (bf16_t)f2.w;
            a.v[12] = (bf16_t)f3.x; a.v[13] = (bf16_t)f3.y;
            a.v[14] = (bf16_t)f3.z; a.v[15] = (bf16_t)f3.w;
        }

#pragma unroll
        for (int tt = 0; tt < 4; ++tt) {
            // b[i] = WT[tt*16+nl][kc+hh+i+(i>=8?8:0)] (table order, matches A)
            const bf16_t* wrow = WT + (size_t)(tt * 16 + nl) * DD + kc + hh;
            Frag16 bfr;
            bfr.h[0] = *(const v8bf*)(wrow);
            bfr.h[1] = *(const v8bf*)(wrow + 16);
            acc[tt] = wmma_bf16(a.v, bfr.v, acc[tt]);
        }
    }

    // Store: C layout = VGPR r -> row (r + half*8), col = tt*16 + nl
#pragma unroll
    for (int tt = 0; tt < 4; ++tt) {
        const int col = tt * 16 + nl;
        const float bval = bias[col];
#pragma unroll
        for (int r = 0; r < 8; ++r) {
            int grow = m0 + r + half * 8;       // global row in [0, B*S)
            int gb   = grow >> 11;              // /S
            int gs   = grow & (SS - 1);
            float val = acc[tt][r] + bval;
            if (z == 2) {
                Vt[((gb * HH_HEADS + h) * DHH + col) * SS + gs] = (bf16_t)val;
            } else {
                bf16_t* P = (z == 0) ? Qp : Kp;
                P[((gb * HH_HEADS + h) * SS + gs) * DHH + col] = (bf16_t)val;
            }
        }
    }
}

// ---------------------------------------------------------------------------
// Kernel 2: flash-style causal attention, one (b,h) per blockIdx.y.
// Block = 256 (8 waves); wave w owns queries [qbase, qbase+16).
// KV chunks of 64; scores and P*V each via 8 WMMAs per chunk.
// K fragments and V fragments load contiguously from global bf16 (V is
// pre-transposed).  P re-shaped C->A layout through per-wave LDS tile.
// Output written bf16 into concat layout [B,S,D].
// ---------------------------------------------------------------------------
__global__ void __launch_bounds__(256)
k_attn(const bf16_t* __restrict__ Qp, const bf16_t* __restrict__ Kp,
       const bf16_t* __restrict__ Vt, const int* __restrict__ maskp,
       bf16_t* __restrict__ Ctx)
{
    const int t    = threadIdx.x;
    const int wave = t >> 5;
    const int lane = t & 31;
    const int nl   = lane & 15;
    const int half = lane >> 4;
    const int hh   = half * 8;
    const int bh   = blockIdx.y;                  // b*H + h
    const int qbase = blockIdx.x * 128 + wave * 16;

    const bf16_t* Qh = Qp + (size_t)bh * (SS * DHH);
    const bf16_t* Kh = Kp + (size_t)bh * (SS * DHH);
    const bf16_t* Vh = Vt + (size_t)bh * (DHH * SS);
    const int mask   = *maskp;

    __shared__ bf16_t Plds[8][16][64];            // per-wave P tile (16x64)

    // Q fragments for both K-chunks of DH=64 (reused across all kv chunks)
    Frag16 qf[2];
    {
        const bf16_t* qrow = Qh + (qbase + nl) * DHH;
#pragma unroll
        for (int kc2 = 0; kc2 < 2; ++kc2) {
            int off = kc2 * 32 + half * 8;
            qf[kc2].h[0] = *(const v8bf*)(qrow + off);
            qf[kc2].h[1] = *(const v8bf*)(qrow + off + 16);
        }
    }

    v8f acc[4];
    float mrun[8], Lsum[8];
#pragma unroll
    for (int tt = 0; tt < 4; ++tt) zero8(acc[tt]);
#pragma unroll
    for (int r = 0; r < 8; ++r) { mrun[r] = -3.0e38f; Lsum[r] = 0.0f; }

    const int nch = mask ? ((qbase + 15) >> 6) + 1 : (SS >> 6);

    for (int ch = 0; ch < nch; ++ch) {
        const int kv0 = ch * 64;

        // ---- scores = (Q * K^T) / 8 ----
        v8f sc[4];
#pragma unroll
        for (int tt = 0; tt < 4; ++tt) zero8(sc[tt]);

#pragma unroll
        for (int kc2 = 0; kc2 < 2; ++kc2) {
#pragma unroll
            for (int tt = 0; tt < 4; ++tt) {
                const bf16_t* krow =
                    Kh + (kv0 + tt * 16 + nl) * DHH + kc2 * 32 + hh;
                Frag16 kb;
                kb.h[0] = *(const v8bf*)(krow);
                kb.h[1] = *(const v8bf*)(krow + 16);
                sc[tt] = wmma_bf16(qf[kc2].v, kb.v, sc[tt]);
            }
        }

        // scale by 1/sqrt(DH) and causal mask (elementwise, exact mapping)
#pragma unroll
        for (int tt = 0; tt < 4; ++tt) {
#pragma unroll
            for (int r = 0; r < 8; ++r) {
                float sval = sc[tt][r] * 0.125f;
                if (mask) {
                    int kidx = kv0 + tt * 16 + nl;
                    int qidx = qbase + r + half * 8;
                    if (kidx > qidx) sval = -1.0e30f;
                }
                sc[tt][r] = sval;
            }
        }

        // ---- online softmax update (row m lives in one 16-lane half) ----
#pragma unroll
        for (int r = 0; r < 8; ++r) {
            float mx = fmaxf(fmaxf(sc[0][r], sc[1][r]),
                             fmaxf(sc[2][r], sc[3][r]));
#pragma unroll
            for (int m2 = 1; m2 < 16; m2 <<= 1)
                mx = fmaxf(mx, __shfl_xor(mx, m2, 32));
            float mnew  = fmaxf(mrun[r], mx);
            float alpha = __expf(mrun[r] - mnew);
            mrun[r] = mnew;
            Lsum[r] *= alpha;
#pragma unroll
            for (int tt = 0; tt < 4; ++tt) acc[tt][r] *= alpha;

            float psum = 0.0f;
#pragma unroll
            for (int tt = 0; tt < 4; ++tt) {
                float p = __expf(sc[tt][r] - mnew);
                psum += p;
                Plds[wave][r + half * 8][tt * 16 + nl] = (bf16_t)p;
            }
#pragma unroll
            for (int m2 = 1; m2 < 16; m2 <<= 1)
                psum += __shfl_xor(psum, m2, 32);
            Lsum[r] += psum;
        }

        // ---- ctx += P * V  (P from LDS as A-fragments; V transposed) ----
        const bf16_t* prow = &Plds[wave][nl][0];
#pragma unroll
        for (int kc2 = 0; kc2 < 2; ++kc2) {
            Frag16 pa;
            int off = kc2 * 32 + half * 8;
            pa.h[0] = *(const v8bf*)(prow + off);
            pa.h[1] = *(const v8bf*)(prow + off + 16);
#pragma unroll
            for (int tt = 0; tt < 4; ++tt) {
                const bf16_t* vcol =
                    Vh + (tt * 16 + nl) * SS + kv0 + kc2 * 32 + hh;
                Frag16 vb;
                vb.h[0] = *(const v8bf*)(vcol);
                vb.h[1] = *(const v8bf*)(vcol + 16);
                acc[tt] = wmma_bf16(pa.v, vb.v, acc[tt]);
            }
        }
    }

    // ---- finalize: ctx / L, write concat layout [B,S,D] (bf16) ----
    const int b = bh >> 4;
    const int h = bh & (HH_HEADS - 1);
#pragma unroll
    for (int r = 0; r < 8; ++r) {
        float inv = 1.0f / Lsum[r];
        int q = qbase + r + half * 8;
#pragma unroll
        for (int tt = 0; tt < 4; ++tt) {
            Ctx[(size_t)(b * SS + q) * DD + h * DHH + tt * 16 + nl] =
                (bf16_t)(acc[tt][r] * inv);
        }
    }
}

// ---------------------------------------------------------------------------
// Kernel 3: output projection  out = Ctx(bf16) @ Wo + bo   (fp32 out)
// grid = (rowTiles=32, colTiles=16), block 256 (8 waves, 16x64 tile each).
// B-fragments load contiguously from bf16 transposed WoT[n][k].
// ---------------------------------------------------------------------------
__global__ void __launch_bounds__(256)
k_oproj(const bf16_t* __restrict__ Ctx, const bf16_t* __restrict__ WoT,
        const float* __restrict__ bo, float* __restrict__ out)
{
    const int t    = threadIdx.x;
    const int wave = t >> 5;
    const int lane = t & 31;
    const int nl   = lane & 15;
    const int half = lane >> 4;
    const int hh   = half * 8;
    const int m0   = blockIdx.x * 128 + wave * 16;
    const int n0   = blockIdx.y * 64;

    v8f acc[4];
#pragma unroll
    for (int tt = 0; tt < 4; ++tt) zero8(acc[tt]);

    const bf16_t* arow = Ctx + (size_t)(m0 + nl) * DD;

#pragma unroll 2
    for (int kc = 0; kc < DD; kc += 32) {
        Frag16 a;
        {
            int off = kc + half * 8;
            a.h[0] = *(const v8bf*)(arow + off);
            a.h[1] = *(const v8bf*)(arow + off + 16);
        }
#pragma unroll
        for (int tt = 0; tt < 4; ++tt) {
            const bf16_t* wrow =
                WoT + (size_t)(n0 + tt * 16 + nl) * DD + kc + hh;
            Frag16 bfr;
            bfr.h[0] = *(const v8bf*)(wrow);
            bfr.h[1] = *(const v8bf*)(wrow + 16);
            acc[tt] = wmma_bf16(a.v, bfr.v, acc[tt]);
        }
    }

#pragma unroll
    for (int tt = 0; tt < 4; ++tt) {
        const int col = n0 + tt * 16 + nl;
        const float bval = bo[col];
#pragma unroll
        for (int r = 0; r < 8; ++r) {
            int grow = m0 + r + half * 8;
            out[(size_t)grow * DD + col] = acc[tt][r] + bval;
        }
    }
}

// ---------------------------------------------------------------------------
// Launcher
// Inputs: 0 query, 1 key, 2 value, 3 Wq, 4 bq, 5 Wk, 6 bk, 7 Wv, 8 bv,
//         9 Wo, 10 bo, 11 look_ahead_mask (int)
// ---------------------------------------------------------------------------
extern "C" void kernel_launch(void* const* d_in, const int* in_sizes, int n_in,
                              void* d_out, int out_size, void* d_ws,
                              size_t ws_size, hipStream_t stream)
{
    (void)in_sizes; (void)n_in; (void)out_size; (void)ws_size;

    const float* q  = (const float*)d_in[0];
    const float* k  = (const float*)d_in[1];
    const float* v  = (const float*)d_in[2];
    const float* Wq = (const float*)d_in[3];
    const float* bq = (const float*)d_in[4];
    const float* Wk = (const float*)d_in[5];
    const float* bk = (const float*)d_in[6];
    const float* Wv = (const float*)d_in[7];
    const float* bv = (const float*)d_in[8];
    const float* Wo = (const float*)d_in[9];
    const float* bo = (const float*)d_in[10];
    const int* maskp = (const int*)d_in[11];
    float* out = (float*)d_out;

    const size_t actAll = (size_t)BB * HH_HEADS * SS * DHH;   // 4,194,304
    const size_t wHead  = (size_t)HH_HEADS * DD * DHH;        // 1,048,576
    bf16_t* Qp  = (bf16_t*)d_ws;
    bf16_t* Kp  = Qp + actAll;
    bf16_t* Vt  = Kp + actAll;
    bf16_t* Ctx = Vt + actAll;
    bf16_t* WqT = Ctx + actAll;
    bf16_t* WkT = WqT + wHead;
    bf16_t* WvT = WkT + wHead;
    bf16_t* WoT = WvT + wHead;            // total 40 MB of workspace

    // 0) weights -> bf16, transposed into B-operand-friendly layouts
    //    per-head [D,DH] -> [DH,D]  (grid.z = head)
    k_transpose_cvt<<<dim3(DHH / 32, DD / 32, HH_HEADS), 256, 0, stream>>>(
        Wq, WqT, DD, DHH);
    k_transpose_cvt<<<dim3(DHH / 32, DD / 32, HH_HEADS), 256, 0, stream>>>(
        Wk, WkT, DD, DHH);
    k_transpose_cvt<<<dim3(DHH / 32, DD / 32, HH_HEADS), 256, 0, stream>>>(
        Wv, WvT, DD, DHH);
    //    Wo [D,D] -> [D,D] transposed
    k_transpose_cvt<<<dim3(DD / 32, DD / 32, 1), 256, 0, stream>>>(
        Wo, WoT, DD, DD);

    // 1) QKV projections (V transposed on store)
    k_qkv_proj<<<dim3(BB * SS / 128, HH_HEADS, 3), 256, 0, stream>>>(
        q, k, v, WqT, WkT, WvT, bq, bk, bv, Qp, Kp, Vt);

    // 2) causal flash attention per (b,h)
    k_attn<<<dim3(SS / 128, BB * HH_HEADS), 256, 0, stream>>>(
        Qp, Kp, Vt, maskp, Ctx);

    // 3) output projection
    k_oproj<<<dim3(BB * SS / 128, DD / 64), 256, 0, stream>>>(
        Ctx, Wo ? WoT : WoT, bo, out);
}
